// Block_40853728920164
// MI455X (gfx1250) — matrix operand order repaired and checked by reading, not compiled, not run on hardware
//
#include <hip/hip_runtime.h>

typedef unsigned short u16;
typedef __attribute__((ext_vector_type(16))) __bf16 v16bf;
typedef __attribute__((ext_vector_type(8)))  float  v8f;

struct alignas(16) U4 { unsigned x, y, z, w; };
union FragB { v16bf v; U4 q[2]; u16 h[16]; };

__device__ __forceinline__ u16 f2bf(float f) {
  unsigned u = __float_as_uint(f);
  u += 0x7FFFu + ((u >> 16) & 1u);       // round-to-nearest-even
  return (u16)(u >> 16);
}

__device__ __forceinline__ float bcast_lane(float x, int src) {
  return __int_as_float(__builtin_amdgcn_ds_bpermute(src << 2, __float_as_int(x)));
}

__device__ __forceinline__ float gelu_f(float x) {
  float x3 = x * x * x;
  return 0.5f * x * (1.f + tanhf(0.7978845608028654f * (x + 0.044715f * x3)));
}

// ---------------------------------------------------------------------------
// fp32 [K][N]  ->  bf16 [N][K]   (weights pre-transpose for contiguous B frags)
// ---------------------------------------------------------------------------
__global__ __launch_bounds__(256)
void transpose_f32_to_bf16(const float* __restrict__ in, u16* __restrict__ out,
                           int K, int N) {
  __shared__ float tile[32][33];
  const int tx = threadIdx.x, ty = threadIdx.y;          // block (32, 8)
  const int k0 = blockIdx.y * 32, n0 = blockIdx.x * 32;
  #pragma unroll
  for (int r = ty; r < 32; r += 8)
    tile[r][tx] = in[(size_t)(k0 + r) * N + (n0 + tx)];
  __syncthreads();
  #pragma unroll
  for (int r = ty; r < 32; r += 8)
    out[(size_t)(n0 + r) * K + (k0 + tx)] = f2bf(tile[tx][r]);
}

// ---------------------------------------------------------------------------
// GroupNorm per (batch, group) over (T=2048, 32 channels); out bf16 [B*T, 1024]
// ---------------------------------------------------------------------------
__global__ __launch_bounds__(256)
void groupnorm_bf16(const float* __restrict__ x, const float* __restrict__ w,
                    const float* __restrict__ b, u16* __restrict__ out) {
  __shared__ float s1[256], s2[256];
  __shared__ float s_mean, s_inv;
  const int tid = threadIdx.x;
  const int bb = blockIdx.x >> 5, g = blockIdx.x & 31;   // 64 blocks = B*32
  float sum = 0.f, sq = 0.f;
  for (int i = tid; i < 2048 * 32; i += 256) {
    int t = i >> 5, c = i & 31;
    float v = x[((size_t)(bb * 2048 + t)) * 1024 + g * 32 + c];
    sum += v; sq += v * v;
  }
  s1[tid] = sum; s2[tid] = sq;
  __syncthreads();
  for (int s = 128; s > 0; s >>= 1) {
    if (tid < s) { s1[tid] += s1[tid + s]; s2[tid] += s2[tid + s]; }
    __syncthreads();
  }
  if (tid == 0) {
    float mean = s1[0] * (1.f / 65536.f);
    float var  = s2[0] * (1.f / 65536.f) - mean * mean;
    s_mean = mean; s_inv = rsqrtf(var + 1e-5f);
  }
  __syncthreads();
  const float mean = s_mean, inv = s_inv;
  for (int i = tid; i < 2048 * 32; i += 256) {
    int t = i >> 5, c = i & 31;
    int ch = g * 32 + c;
    size_t idx = ((size_t)(bb * 2048 + t)) * 1024 + ch;
    out[idx] = f2bf((x[idx] - mean) * inv * w[ch] + b[ch]);
  }
}

// ---------------------------------------------------------------------------
// bf16 WMMA GEMM:  C[M,N] = A[M,K] * Bt[N,K]^T + bias  (+mode epilogue)
// 256 threads = 8 waves (2x4); block tile 128x128; wave tile 64x32.
// MODE 0: QKV split  (q/k bf16, v transposed bf16, k/v fp32 -> present)
// MODE 1: +resid, fp32 out        (attention projection)
// MODE 2: gelu, bf16 out          (fc)
// MODE 3: +resid, fp32 out        (mlp projection -> final x)
// ---------------------------------------------------------------------------
template<int MODE>
__global__ __launch_bounds__(256)
void gemm_bf16(const u16* __restrict__ A, const u16* __restrict__ Bt,
               const float* __restrict__ bias, const float* __restrict__ resid,
               float* __restrict__ outF, u16* __restrict__ outB,
               u16* __restrict__ outB2, u16* __restrict__ outB3,
               int M, int N, int K) {
  const int tid = threadIdx.x;
  const int lane = tid & 31, wid = tid >> 5;
  const int l15 = lane & 15, hi = lane >> 4;
  const int wm = wid >> 2, wn = wid & 3;
  const int m0 = blockIdx.y * 128 + wm * 64;
  const int n0 = blockIdx.x * 128 + wn * 32;
  (void)M;

  const v8f zero = {0.f, 0.f, 0.f, 0.f, 0.f, 0.f, 0.f, 0.f};
  v8f acc[4][2];
  #pragma unroll
  for (int i = 0; i < 4; i++)
    #pragma unroll
    for (int j = 0; j < 2; j++) acc[i][j] = zero;

  for (int k0 = 0; k0 < K; k0 += 32) {
    FragB a[4], b[2];
    #pragma unroll
    for (int i = 0; i < 4; i++) {            // A frag: rows striped by lane
      const u16* p = A + (size_t)(m0 + i * 16 + l15) * K + k0 + hi * 8;
      a[i].q[0] = *reinterpret_cast<const U4*>(p);
      a[i].q[1] = *reinterpret_cast<const U4*>(p + 16);
    }
    #pragma unroll
    for (int j = 0; j < 2; j++) {            // B frag: K 8h..8h+7 and 16+8h..16+8h+7 per lane
      const u16* p = Bt + (size_t)(n0 + j * 16 + l15) * K + k0 + hi * 8;
      b[j].q[0] = *reinterpret_cast<const U4*>(p);
      b[j].q[1] = *reinterpret_cast<const U4*>(p + 16);
    }
    #pragma unroll
    for (int i = 0; i < 4; i++)
      #pragma unroll
      for (int j = 0; j < 2; j++)
        acc[i][j] = __builtin_amdgcn_wmma_f32_16x16x32_bf16(
            false, a[i].v, false, b[j].v, (short)0, acc[i][j], false, false);
  }

  #pragma unroll
  for (int i = 0; i < 4; i++) {
    #pragma unroll
    for (int j = 0; j < 2; j++) {
      const int c = n0 + j * 16 + l15;
      const float bc = bias[c];
      #pragma unroll
      for (int v = 0; v < 8; v++) {
        const int r = m0 + i * 16 + v + 8 * hi;
        float val = acc[i][j][v] + bc;
        if constexpr (MODE == 0) {
          int b_ = r >> 11, t = r & 2047;
          int part = c >> 10, cc = c & 1023;
          int h = cc >> 6, d = cc & 63;
          size_t idx = ((size_t)(b_ * 16 + h) * 2048 + t) * 64 + d;
          if (part == 0) {
            outB[idx] = f2bf(val);                                   // q
          } else if (part == 1) {
            outB2[idx] = f2bf(val);                                  // k bf16
            outF[idx] = val;                                         // present k
          } else {
            outB3[((size_t)(b_ * 16 + h) * 64 + d) * 2048 + t] = f2bf(val); // v^T
            outF[4194304 + idx] = val;                               // present v
          }
        } else if constexpr (MODE == 1) {
          outF[(size_t)r * N + c] = val + resid[(size_t)r * N + c];
        } else if constexpr (MODE == 2) {
          outB[(size_t)r * N + c] = f2bf(gelu_f(val));
        } else {
          outF[(size_t)r * N + c] = val + resid[(size_t)r * N + c];
        }
      }
    }
  }
}

// ---------------------------------------------------------------------------
// Flash attention (causal, no 1/sqrt(d) scale). One wave = 16 query rows.
// S^T = K * Q^T via WMMA: the two 16-key C tiles of a 32-key chunk are
// exactly the A-fragment layout for the P*V WMMA. V is pre-transposed.
// ---------------------------------------------------------------------------
__global__ __launch_bounds__(256)
void attn_flash(const u16* __restrict__ qb, const u16* __restrict__ kb,
                const u16* __restrict__ vt, u16* __restrict__ ab) {
  const int lane = threadIdx.x & 31, wid = threadIdx.x >> 5;
  const int l15 = lane & 15, hi = lane >> 4;
  const int bh = blockIdx.y;                       // b*16 + head
  const int q0 = blockIdx.x * 128 + wid * 16;
  const u16* Q  = qb + (size_t)bh * 2048 * 64;
  const u16* Kp = kb + (size_t)bh * 2048 * 64;
  const u16* Vp = vt + (size_t)bh * 64 * 2048;

  FragB qf[2];                                     // Q^T as B operand
  #pragma unroll
  for (int c = 0; c < 2; c++) {
    const u16* p = Q + (size_t)(q0 + l15) * 64 + c * 32 + hi * 8;
    qf[c].q[0] = *reinterpret_cast<const U4*>(p);
    qf[c].q[1] = *reinterpret_cast<const U4*>(p + 16);
  }

  const v8f zero = {0.f, 0.f, 0.f, 0.f, 0.f, 0.f, 0.f, 0.f};
  v8f O[4] = {zero, zero, zero, zero};
  float m_run = -1e30f, l_run = 0.f;
  const int qidx = q0 + l15;
  const int nkc = ((q0 + 15) >> 5) + 1;

  for (int kc = 0; kc < nkc; kc++) {
    const int kbase = kc << 5;
    v8f st[2] = {zero, zero};
    #pragma unroll
    for (int s = 0; s < 2; s++) {
      #pragma unroll
      for (int c = 0; c < 2; c++) {
        FragB kf;                                  // K rows as A operand
        const u16* p = Kp + (size_t)(kbase + s * 16 + l15) * 64 + c * 32 + hi * 8;
        kf.q[0] = *reinterpret_cast<const U4*>(p);
        kf.q[1] = *reinterpret_cast<const U4*>(p + 16);
        st[s] = __builtin_amdgcn_wmma_f32_16x16x32_bf16(
            false, kf.v, false, qf[c].v, (short)0, st[s], false, false);
      }
    }
    // causal mask + chunk max (lane holds 16 keys; partner lane the other 16)
    float mloc = -1e30f;
    #pragma unroll
    for (int s = 0; s < 2; s++)
      #pragma unroll
      for (int v = 0; v < 8; v++) {
        int key = kbase + s * 16 + v + 8 * hi;
        float e = (key <= qidx) ? st[s][v] : -1e30f;
        st[s][v] = e;
        mloc = fmaxf(mloc, e);
      }
    mloc = fmaxf(mloc, bcast_lane(mloc, lane ^ 16));
    const float m_new = fmaxf(m_run, mloc);
    const float scale = __expf(m_run - m_new);
    float lloc = 0.f;
    FragB pf;                                      // P in A-fragment layout
    #pragma unroll
    for (int s = 0; s < 2; s++)
      #pragma unroll
      for (int v = 0; v < 8; v++) {
        float p = __expf(st[s][v] - m_new);
        lloc += p;
        pf.h[s * 8 + v] = f2bf(p);
      }
    lloc += bcast_lane(lloc, lane ^ 16);
    l_run = l_run * scale + lloc;
    m_run = m_new;
    #pragma unroll
    for (int v = 0; v < 8; v++) {                  // rescale O rows
      float sc = bcast_lane(scale, v + 8 * hi);
      #pragma unroll
      for (int j = 0; j < 4; j++) O[j][v] *= sc;
    }
    #pragma unroll
    for (int j = 0; j < 4; j++) {                  // O += P * V
      FragB vf;                                    // V^T rows: keys 8h..8h+7, 16+8h..16+8h+7
      const u16* p = Vp + (size_t)(j * 16 + l15) * 2048 + kbase + hi * 8;
      vf.q[0] = *reinterpret_cast<const U4*>(p);
      vf.q[1] = *reinterpret_cast<const U4*>(p + 16);
      O[j] = __builtin_amdgcn_wmma_f32_16x16x32_bf16(
          false, pf.v, false, vf.v, (short)0, O[j], false, false);
    }
  }

  const float linv = 1.f / l_run;
  const int b_ = bh >> 4, h = bh & 15;
  #pragma unroll
  for (int v = 0; v < 8; v++) {
    float lv = bcast_lane(linv, v + 8 * hi);
    int r = q0 + v + 8 * hi;
    #pragma unroll
    for (int j = 0; j < 4; j++) {
      size_t idx = ((size_t)(b_ * 2048 + r)) * 1024 + h * 64 + j * 16 + l15;
      ab[idx] = f2bf(O[j][v] * lv);
    }
  }
}

// ---------------------------------------------------------------------------
extern "C" void kernel_launch(void* const* d_in, const int* in_sizes, int n_in,
                              void* d_out, int out_size, void* d_ws, size_t ws_size,
                              hipStream_t stream) {
  (void)in_sizes; (void)n_in; (void)out_size; (void)ws_size;
  const float* x       = (const float*)d_in[0];
  const float* w_attn  = (const float*)d_in[1];
  const float* b_attn  = (const float*)d_in[2];
  const float* w_aproj = (const float*)d_in[3];
  const float* b_aproj = (const float*)d_in[4];
  const float* ln1_w   = (const float*)d_in[5];
  const float* ln1_b   = (const float*)d_in[6];
  const float* ln2_w   = (const float*)d_in[7];
  const float* ln2_b   = (const float*)d_in[8];
  const float* w_fc    = (const float*)d_in[9];
  const float* b_fc    = (const float*)d_in[10];
  const float* w_mproj = (const float*)d_in[11];
  const float* b_mproj = (const float*)d_in[12];

  float* out_x   = (float*)d_out;
  float* present = out_x + 4194304;              // [2,B,nh,T,hd] fp32

  char* w = (char*)d_ws;
  u16* xn      = (u16*)w;  w += 8388608;         // gn1(x) bf16 [4096,1024]
  u16* wattnT  = (u16*)w;  w += 6291456;         // [3072,1024] bf16
  u16* waprojT = (u16*)w;  w += 2097152;         // [1024,1024] bf16
  u16* wfcT    = (u16*)w;  w += 8388608;         // [4096,1024] bf16
  u16* wmprojT = (u16*)w;  w += 8388608;         // [1024,4096] bf16
  u16* qbuf    = (u16*)w;  w += 8388608;         // [32][2048][64] bf16
  u16* kbuf    = (u16*)w;  w += 8388608;         // [32][2048][64] bf16
  u16* vtbuf   = (u16*)w;  w += 8388608;         // [32][64][2048] bf16
  u16* abuf    = (u16*)w;  w += 8388608;         // attn out bf16 [4096,1024]
  float* x1    = (float*)w; w += 16777216;       // x + attn fp32
  u16* x2n     = (u16*)w;  w += 8388608;         // gn2(x1) bf16
  u16* hbuf    = (u16*)w;  w += 33554432;        // gelu(fc) bf16 [4096,4096]

  dim3 tb(32, 8);
  transpose_f32_to_bf16<<<dim3(96, 32), tb, 0, stream>>>(w_attn, wattnT, 1024, 3072);
  transpose_f32_to_bf16<<<dim3(32, 32), tb, 0, stream>>>(w_aproj, waprojT, 1024, 1024);
  transpose_f32_to_bf16<<<dim3(128, 32), tb, 0, stream>>>(w_fc, wfcT, 1024, 4096);
  transpose_f32_to_bf16<<<dim3(32, 128), tb, 0, stream>>>(w_mproj, wmprojT, 4096, 1024);

  groupnorm_bf16<<<64, 256, 0, stream>>>(x, ln1_w, ln1_b, xn);

  gemm_bf16<0><<<dim3(24, 32), 256, 0, stream>>>(
      xn, wattnT, b_attn, nullptr, present, qbuf, kbuf, vtbuf, 4096, 3072, 1024);

  attn_flash<<<dim3(16, 32), 256, 0, stream>>>(qbuf, kbuf, vtbuf, abuf);

  gemm_bf16<1><<<dim3(8, 32), 256, 0, stream>>>(
      abuf, waprojT, b_aproj, x, x1, nullptr, nullptr, nullptr, 4096, 1024, 1024);

  groupnorm_bf16<<<64, 256, 0, stream>>>(x1, ln2_w, ln2_b, x2n);

  gemm_bf16<2><<<dim3(32, 32), 256, 0, stream>>>(
      x2n, wfcT, b_fc, nullptr, nullptr, hbuf, nullptr, nullptr, 4096, 4096, 1024);

  gemm_bf16<3><<<dim3(8, 32), 256, 0, stream>>>(
      hbuf, wmprojT, b_mproj, x1, out_x, nullptr, nullptr, nullptr, 4096, 1024, 4096);
}
